// IthemalRNN_31018253812272
// MI455X (gfx1250) — compile-verified, matrix operand order checked by reading, not run on hardware
//
#include <hip/hip_runtime.h>
#include <hip/hip_bf16.h>

// ---------------------------------------------------------------------------
// Ithemal RNN on MI455X (gfx1250): bf16 WMMA token-LSTM + WMMA input-projection
// for the instruction-LSTM, single-workgroup sequential recurrence for the
// 4096-step scan.
// ---------------------------------------------------------------------------

typedef __bf16 v8bf  __attribute__((ext_vector_type(8)));
typedef __bf16 v16bf __attribute__((ext_vector_type(16)));
typedef float  v8f   __attribute__((ext_vector_type(8)));

#define SHUF16(lo, hi) __builtin_shufflevector((lo), (hi), 0,1,2,3,4,5,6,7,8,9,10,11,12,13,14,15)

#define NI   4096   // instructions
#define TT   16     // max tokens
#define EE   256    // embedding
#define HH   512    // hidden
#define GG   2048   // 4*H gates
#define KTOK 768    // E + H

__device__ __forceinline__ float sigm(float x) { return 1.0f / (1.0f + __expf(-x)); }

// ---------------------------------------------------------------------------
// Prep kernels
// ---------------------------------------------------------------------------

// Zero H double-buffer + C (contiguous region), one float4 per thread.
__global__ void zero_kernel(float4* __restrict__ p, int n4) {
  int i = blockIdx.x * 256 + threadIdx.x;
  if (i < n4) p[i] = make_float4(0.f, 0.f, 0.f, 0.f);
}

// Pack token-LSTM B operand: B[k][n] = Wtok[n][k], k<256 from w_ih, else w_hh.
// Fragment-ordered blob: ((kt*128 + nt)*32 + lane)*16 + i, with
//   K = kt*32 + (lane<16?0:8) + (i<8? i : i+8),  N = nt*16 + (lane&15)
__global__ void pack_btok_kernel(const float* __restrict__ w_ih,
                                 const float* __restrict__ w_hh,
                                 __bf16* __restrict__ blob) {
  int idx = blockIdx.x * 256 + threadIdx.x;          // < 24*128*512
  int i    = idx & 15;
  int lane = (idx >> 4) & 31;
  int nt   = (idx >> 9) & 127;
  int kt   = idx >> 16;                              // 0..23
  int K = kt * 32 + ((lane < 16) ? 0 : 8) + ((i < 8) ? i : i + 8);
  int Nc = nt * 16 + (lane & 15);
  float v = (K < EE) ? w_ih[Nc * EE + K] : w_hh[Nc * HH + (K - EE)];
  blob[idx] = (__bf16)v;
}

// Pack w_ih_ins [2048,512] the same way (16 k-tiles).
__global__ void pack_bins_kernel(const float* __restrict__ w_ih,
                                 __bf16* __restrict__ blob) {
  int idx = blockIdx.x * 256 + threadIdx.x;          // < 16*128*512
  int i    = idx & 15;
  int lane = (idx >> 4) & 31;
  int nt   = (idx >> 9) & 127;
  int kt   = idx >> 16;                              // 0..15
  int K = kt * 32 + ((lane < 16) ? 0 : 8) + ((i < 8) ? i : i + 8);
  int Nc = nt * 16 + (lane & 15);
  blob[idx] = (__bf16)w_ih[Nc * HH + K];
}

__global__ void bias_kernel(const float* __restrict__ bi_t, const float* __restrict__ bh_t,
                            const float* __restrict__ bi_i, const float* __restrict__ bh_i,
                            float* __restrict__ btok, float* __restrict__ bins) {
  int i = blockIdx.x * 256 + threadIdx.x;
  if (i < GG) { btok[i] = bi_t[i] + bh_t[i]; bins[i] = bi_i[i] + bh_i[i]; }
}

// WT[k*2048 + j] = w_hh_ins[j*512 + k]  (k-major for coalesced GEMV reads)
__global__ void transpose_whh_kernel(const float* __restrict__ whh, float* __restrict__ wt) {
  int idx = blockIdx.x * 256 + threadIdx.x;          // < 512*2048
  int k = idx >> 11, j = idx & 2047;
  wt[idx] = whh[j * HH + k];
}

// ---------------------------------------------------------------------------
// Token-LSTM fused step: gates GEMM (WMMA, K=768) + cell + length masking.
// grid = (256 M-blocks, 4 gate-column groups), block = 256 (8 waves).
// Wave w owns 4 N-tiles; tile tt = w*4+j -> gate chunk q = tt>>3, local lt = tt&7,
// global gate column = q*512 + cg*128 + lt*16 + (lane&15).
// ---------------------------------------------------------------------------
__global__ __launch_bounds__(256) void tok_step_kernel(
    int t,
    const int*   __restrict__ token_ids,
    const int*   __restrict__ lengths,
    const float* __restrict__ emb,
    const __bf16* __restrict__ bfrag,
    const float* __restrict__ btok,
    const __bf16* __restrict__ Hrd,
    __bf16*       __restrict__ Hwr,
    float*        __restrict__ Cst)
{
  __shared__ __bf16 As[16 * KTOK];   // [row][k]  (x | h), 24 KB
  __shared__ float  Gs[16 * 512];    // [row][q*128 + u_local], 32 KB

  const int tid  = threadIdx.x;
  const int mblk = blockIdx.x;       // 16-row block
  const int cg   = blockIdx.y;       // gate-column group (128 units)

  // Stage A: embedding gather (f32->bf16) and previous h (bf16 passthrough).
  for (int idx = tid; idx < 16 * KTOK; idx += 256) {
    int r = idx / KTOK, k = idx - r * KTOK;
    int n = mblk * 16 + r;
    if (k < EE) {
      int tok = token_ids[n * TT + t];
      As[idx] = (__bf16)emb[tok * EE + k];
    } else {
      As[idx] = Hrd[(size_t)n * HH + (k - EE)];
    }
  }
  __syncthreads();

  const int w = tid >> 5, lane = tid & 31;
  const int lrow = lane & 15, hi = lane >> 4;

  v8f acc[4];
  #pragma unroll
  for (int j = 0; j < 4; ++j) {
    int tt = w * 4 + j, q = tt >> 3, lt = tt & 7;
    float b = btok[q * 512 + cg * 128 + lt * 16 + lrow];
    v8f a = {b, b, b, b, b, b, b, b};
    acc[j] = a;
  }

  for (int kt = 0; kt < 24; ++kt) {
    // A fragment: row = lane&15, K chunks {base..base+7} and {base+16..base+23}
    const v8bf* ap = (const v8bf*)&As[lrow * KTOK + kt * 32 + hi * 8];
    v16bf afrag = SHUF16(ap[0], ap[2]);
    #pragma unroll
    for (int j = 0; j < 4; ++j) {
      int tt = w * 4 + j, q = tt >> 3, lt = tt & 7;
      int ntg = q * 32 + cg * 8 + lt;
      const v8bf* bp = (const v8bf*)&bfrag[((size_t)(kt * 128 + ntg)) * 512 + lane * 16];
      v16bf bb = SHUF16(bp[0], bp[1]);
      acc[j] = __builtin_amdgcn_wmma_f32_16x16x32_bf16(false, afrag, false, bb,
                                                       (short)0, acc[j], false, false);
    }
  }

  // Spill gates to LDS (C/D layout: vgpr r -> M = r + 8*hi, N = lane&15).
  #pragma unroll
  for (int j = 0; j < 4; ++j) {
    int tt = w * 4 + j, q = tt >> 3, lt = tt & 7;
    int col = q * 128 + lt * 16 + lrow;
    #pragma unroll
    for (int r = 0; r < 8; ++r) {
      int row = hi ? (r + 8) : r;
      Gs[row * 512 + col] = acc[j][r];
    }
  }
  __syncthreads();

  // LSTM cell for 16 rows x 128 units; freeze state where t >= lengths[n].
  for (int rep = 0; rep < 8; ++rep) {
    int flat = rep * 256 + tid;            // 0..2047
    int row = flat >> 7, ul = flat & 127;
    int n = mblk * 16 + row;
    size_t gi = (size_t)n * HH + cg * 128 + ul;
    if (t < lengths[n]) {
      float ig = sigm (Gs[row * 512 + 0 * 128 + ul]);
      float fg = sigm (Gs[row * 512 + 1 * 128 + ul]);
      float gg = tanhf(Gs[row * 512 + 2 * 128 + ul]);
      float og = sigm (Gs[row * 512 + 3 * 128 + ul]);
      float c = fg * Cst[gi] + ig * gg;
      Cst[gi] = c;
      Hwr[gi] = (__bf16)(og * tanhf(c));
    } else {
      Hwr[gi] = Hrd[gi];                   // carry frozen h into write buffer
    }
  }
}

// ---------------------------------------------------------------------------
// Gx = h_fin @ w_ih_ins^T + b_ins   ([4096,512] @ [512,2048], WMMA)
// ---------------------------------------------------------------------------
__global__ __launch_bounds__(256) void gx_gemm_kernel(
    const __bf16* __restrict__ Hf,
    const __bf16* __restrict__ bfrag,
    const float*  __restrict__ bins,
    float*        __restrict__ Gx)
{
  __shared__ __bf16 As[16 * HH];
  const int tid = threadIdx.x, mblk = blockIdx.x, cg = blockIdx.y;

  for (int idx = tid; idx < 16 * HH; idx += 256)
    As[idx] = Hf[(size_t)(mblk * 16 + (idx >> 9)) * HH + (idx & 511)];
  __syncthreads();

  const int w = tid >> 5, lane = tid & 31;
  const int lrow = lane & 15, hi = lane >> 4;

  v8f acc[4];
  #pragma unroll
  for (int j = 0; j < 4; ++j) {
    int tt = w * 4 + j, q = tt >> 3, lt = tt & 7;
    float b = bins[q * 512 + cg * 128 + lt * 16 + lrow];
    v8f a = {b, b, b, b, b, b, b, b};
    acc[j] = a;
  }

  for (int kt = 0; kt < 16; ++kt) {
    const v8bf* ap = (const v8bf*)&As[lrow * HH + kt * 32 + hi * 8];
    v16bf afrag = SHUF16(ap[0], ap[2]);
    #pragma unroll
    for (int j = 0; j < 4; ++j) {
      int tt = w * 4 + j, q = tt >> 3, lt = tt & 7;
      int ntg = q * 32 + cg * 8 + lt;
      const v8bf* bp = (const v8bf*)&bfrag[((size_t)(kt * 128 + ntg)) * 512 + lane * 16];
      v16bf bb = SHUF16(bp[0], bp[1]);
      acc[j] = __builtin_amdgcn_wmma_f32_16x16x32_bf16(false, afrag, false, bb,
                                                       (short)0, acc[j], false, false);
    }
  }

  #pragma unroll
  for (int j = 0; j < 4; ++j) {
    int tt = w * 4 + j, q = tt >> 3, lt = tt & 7;
    int ncol = q * 512 + cg * 128 + lt * 16 + lrow;
    #pragma unroll
    for (int r = 0; r < 8; ++r) {
      int row = hi ? (r + 8) : r;
      Gx[(size_t)(mblk * 16 + row) * GG + ncol] = acc[j][r];
    }
  }
}

// ---------------------------------------------------------------------------
// Sequential instruction-LSTM: one workgroup, 1024 threads, 4096 steps.
// gates[j] = Gx[n][j] + sum_k h[k] * WT[k][j];  thread tid owns j=tid, j=tid+1024.
// ---------------------------------------------------------------------------
__global__ __launch_bounds__(1024) void ins_lstm_kernel(
    const float* __restrict__ Gx,
    const float* __restrict__ WT,     // [512][2048] k-major
    const float* __restrict__ wlin,
    const float* __restrict__ blin,
    float* __restrict__ out)
{
  __shared__ float hs[HH];
  __shared__ float gs[GG];
  const int tid = threadIdx.x;

  float c = 0.0f;
  if (tid < HH) hs[tid] = 0.0f;
  __syncthreads();

  for (int n = 0; n < NI; ++n) {
    const float* gx = Gx + (size_t)n * GG;
    float a0 = gx[tid];
    float a1 = gx[GG / 2 + tid];
    #pragma unroll 8
    for (int k = 0; k < HH; ++k) {
      float hk = hs[k];                       // LDS broadcast
      a0 = fmaf(hk, WT[k * GG + tid],            a0);
      a1 = fmaf(hk, WT[k * GG + GG / 2 + tid],   a1);
    }
    gs[tid] = a0;
    gs[GG / 2 + tid] = a1;
    __syncthreads();
    if (tid < HH) {
      float ig = sigm (gs[tid]);
      float fg = sigm (gs[HH + tid]);
      float gg = tanhf(gs[2 * HH + tid]);
      float og = sigm (gs[3 * HH + tid]);
      c = fg * c + ig * gg;
      hs[tid] = og * tanhf(c);
    }
    __syncthreads();
  }

  // linear head: out = h_last . w_lin + b_lin
  if (tid < HH) gs[tid] = hs[tid] * wlin[tid];
  __syncthreads();
  for (int s = 256; s >= 1; s >>= 1) {
    if (tid < s) gs[tid] += gs[tid + s];
    __syncthreads();
  }
  if (tid == 0) out[0] = gs[0] + blin[0];
}

// ---------------------------------------------------------------------------
// Workspace layout (bytes, all offsets 256-aligned; hipMalloc base is aligned)
// ---------------------------------------------------------------------------
static constexpr size_t OFF_BFRAG_TOK = 0;                                  // 24*128*512 bf16
static constexpr size_t SZ_BFRAG_TOK  = 24ull * 128 * 512 * 2;              // 3 MB
static constexpr size_t OFF_BFRAG_INS = OFF_BFRAG_TOK + SZ_BFRAG_TOK;
static constexpr size_t SZ_BFRAG_INS  = 16ull * 128 * 512 * 2;              // 2 MB
static constexpr size_t OFF_BTOK      = OFF_BFRAG_INS + SZ_BFRAG_INS;
static constexpr size_t OFF_BINS      = OFF_BTOK + GG * 4;
static constexpr size_t OFF_HBF0      = OFF_BINS + GG * 4;                  // 4 MB
static constexpr size_t SZ_HBF_ONE    = (size_t)NI * HH * 2;
static constexpr size_t OFF_HBF1      = OFF_HBF0 + SZ_HBF_ONE;              // 4 MB
static constexpr size_t OFF_CELL      = OFF_HBF1 + SZ_HBF_ONE;              // 8 MB
static constexpr size_t SZ_CELL       = (size_t)NI * HH * 4;
static constexpr size_t OFF_WT        = OFF_CELL + SZ_CELL;                 // 4 MB
static constexpr size_t SZ_WT         = (size_t)HH * GG * 4;
static constexpr size_t OFF_GX        = OFF_WT + SZ_WT;                     // 32 MB
static constexpr size_t SZ_GX         = (size_t)NI * GG * 4;

extern "C" void kernel_launch(void* const* d_in, const int* in_sizes, int n_in,
                              void* d_out, int out_size, void* d_ws, size_t ws_size,
                              hipStream_t stream) {
  const int*   token_ids = (const int*)  d_in[0];
  const int*   lengths   = (const int*)  d_in[1];
  const float* embedding = (const float*)d_in[2];
  const float* w_ih_tok  = (const float*)d_in[3];
  const float* w_hh_tok  = (const float*)d_in[4];
  const float* b_ih_tok  = (const float*)d_in[5];
  const float* b_hh_tok  = (const float*)d_in[6];
  const float* w_ih_ins  = (const float*)d_in[7];
  const float* w_hh_ins  = (const float*)d_in[8];
  const float* b_ih_ins  = (const float*)d_in[9];
  const float* b_hh_ins  = (const float*)d_in[10];
  const float* w_lin     = (const float*)d_in[11];
  const float* b_lin     = (const float*)d_in[12];

  char* ws = (char*)d_ws;
  __bf16* bfrag_tok = (__bf16*)(ws + OFF_BFRAG_TOK);
  __bf16* bfrag_ins = (__bf16*)(ws + OFF_BFRAG_INS);
  float*  btok      = (float*) (ws + OFF_BTOK);
  float*  bins      = (float*) (ws + OFF_BINS);
  __bf16* hbf0      = (__bf16*)(ws + OFF_HBF0);
  __bf16* hbf1      = (__bf16*)(ws + OFF_HBF1);
  float*  cst       = (float*) (ws + OFF_CELL);
  float*  wt        = (float*) (ws + OFF_WT);
  float*  gx        = (float*) (ws + OFF_GX);

  // 1) zero H double-buffer + C (16 MB contiguous)
  {
    int n4 = (int)((2 * SZ_HBF_ONE + SZ_CELL) / 16);   // 1,048,576 float4
    zero_kernel<<<dim3((n4 + 255) / 256), dim3(256), 0, stream>>>(
        (float4*)(ws + OFF_HBF0), n4);
  }
  // 2) weight packing / bias / transpose
  pack_btok_kernel<<<dim3(6144), dim3(256), 0, stream>>>(w_ih_tok, w_hh_tok, bfrag_tok);
  pack_bins_kernel<<<dim3(4096), dim3(256), 0, stream>>>(w_ih_ins, bfrag_ins);
  bias_kernel<<<dim3(8), dim3(256), 0, stream>>>(b_ih_tok, b_hh_tok, b_ih_ins, b_hh_ins,
                                                 btok, bins);
  transpose_whh_kernel<<<dim3(4096), dim3(256), 0, stream>>>(w_hh_ins, wt);

  // 3) token-level LSTM, 16 fused WMMA steps, H double-buffered
  for (int t = 0; t < TT; ++t) {
    const __bf16* hrd = (t & 1) ? hbf1 : hbf0;
    __bf16*       hwr = (t & 1) ? hbf0 : hbf1;
    tok_step_kernel<<<dim3(256, 4), dim3(256), 0, stream>>>(
        t, token_ids, lengths, embedding, bfrag_tok, btok, hrd, hwr, cst);
  }
  // after t=15 the final h_fin lives in hbf0

  // 4) instruction-LSTM input projection via WMMA
  gx_gemm_kernel<<<dim3(256, 4), dim3(256), 0, stream>>>(hbf0, bfrag_ins, bins, gx);

  // 5) sequential 4096-step recurrence + head
  ins_lstm_kernel<<<dim3(1), dim3(1024), 0, stream>>>(gx, wt, w_lin, b_lin, (float*)d_out);
}
